// BNNEnsemble_87093346828975
// MI455X (gfx1250) — matrix operand order repaired and checked by reading, not compiled, not run on hardware
//
#include <hip/hip_runtime.h>
#include <hip/hip_bf16.h>
#include <math.h>

typedef __bf16 bf16_t;
typedef __attribute__((ext_vector_type(16))) __bf16 v16bf;
typedef __attribute__((ext_vector_type(8)))  __bf16 v8bf;
typedef __attribute__((ext_vector_type(8)))  float  v8f;

#define DIMO 128
#define DIMU 32
#define DIN  160      // = DIMO + DIMU, 5 K-chunks of 32
#define HDIM 512      // hidden width, 16 K-chunks of 32
#define KOUT 129
#define NREAL 258     // 2*KOUT
#define NPAD 272      // padded to 17 tiles of 16
#define ROWS 64       // batch rows per block
#define NTHREADS 256  // 8 wave32

__device__ __forceinline__ float softplus_f(float x) {
  return (x > 30.0f) ? x : log1pf(expf(x));
}

__device__ __forceinline__ float silu_f(float x) {
  // x * sigmoid(x) with fast v_rcp_f32 instead of IEEE divide
  return x * __builtin_amdgcn_rcpf(1.0f + __expf(-x));
}

// ---------------- prep kernels ----------------

__global__ void normalize_x_kernel(const float* __restrict__ obs, const float* __restrict__ act,
                                   const float* __restrict__ mu, const float* __restrict__ sigma,
                                   bf16_t* __restrict__ xbf, int B) {
  int idx = blockIdx.x * blockDim.x + threadIdx.x;
  if (idx >= B * DIN) return;
  int b = idx / DIN, d = idx % DIN;
  float v = (d < DIMO) ? obs[(size_t)b * DIMO + d] : act[(size_t)b * DIMU + (d - DIMO)];
  xbf[idx] = (bf16_t)((v - mu[d]) / sigma[d]);
}

__global__ void bin_rows_kernel(const int* __restrict__ elite, const int* __restrict__ midx,
                                int* __restrict__ counts, int* __restrict__ rowlist, int B) {
  int b = blockIdx.x * blockDim.x + threadIdx.x;
  if (b >= B) return;
  int sel = elite[midx[b]];
  int pos = atomicAdd(&counts[sel], 1);
  rowlist[(size_t)sel * B + pos] = b;
}

// dst[m][n][k] = (bf16) src[m][k][n]   (zero-pad n >= N up to Npad)
__global__ void transpose_w_kernel(const float* __restrict__ src, bf16_t* __restrict__ dst,
                                   int Kdim, int N, int Npad, long total) {
  long idx = (long)blockIdx.x * blockDim.x + threadIdx.x;
  if (idx >= total) return;
  int  k = (int)(idx % Kdim);
  long t = idx / Kdim;
  int  n = (int)(t % Npad);
  long m = t / Npad;
  float v = (n < N) ? src[(m * Kdim + k) * (long)N + n] : 0.0f;
  dst[idx] = (bf16_t)v;
}

// ---------------- WMMA fragment helpers ----------------
// A (16x32 bf16): lane half selects K sub-offset 0/8; row = lane&15.
// B (32x16 bf16): lane half selects K sub-offset 0/8; col = lane&15.
__device__ __forceinline__ v16bf load_frag(const bf16_t* p0, const bf16_t* p1) {
  v8bf x0 = *(const v8bf*)p0;
  v8bf x1 = *(const v8bf*)p1;
  v16bf f;
#pragma unroll
  for (int j = 0; j < 8; ++j) { f[j] = x0[j]; f[j + 8] = x1[j]; }
  return f;
}

// ---------------- fused MLP layers ----------------
// Hidden layer: 64 rows x 512 cols. Each wave owns a 64x64 output block:
// 4 row-subtiles x 4 col-subtiles, K-outer loop, 16 live f32 accumulators.
// Per K-chunk: 4 A frags (LDS), 4 B frags (global), 16 WMMAs.
template <int KD>
__device__ __forceinline__ void gemm_hidden(const bf16_t* __restrict__ inLds,
                                            const bf16_t* __restrict__ Wt,
                                            const float* __restrict__ bias,
                                            bf16_t* __restrict__ outLds) {
  const int lane  = threadIdx.x & 31;
  const int wave  = threadIdx.x >> 5;
  const int l15   = lane & 15;
  const int aoff  = (lane >> 4) * 8;
  const int boff  = (lane >> 4) * 8;
  const int rhalf = (lane >> 4) * 8;
  const int cbase = wave * 64;  // this wave's first output column

  v8f acc[4][4];
#pragma unroll
  for (int c = 0; c < 4; ++c)
#pragma unroll
    for (int r = 0; r < 4; ++r)
#pragma unroll
      for (int i = 0; i < 8; ++i) acc[c][r][i] = 0.0f;

  const bf16_t* aBase = inLds + (size_t)l15 * KD;
  const bf16_t* bBase = Wt + (size_t)(cbase + l15) * KD;

  for (int k0 = 0; k0 < KD; k0 += 32) {
    v16bf a[4];
#pragma unroll
    for (int r = 0; r < 4; ++r) {
      const bf16_t* p = aBase + (size_t)(r * 16) * KD + k0 + aoff;
      a[r] = load_frag(p, p + 16);
    }
#pragma unroll
    for (int c = 0; c < 4; ++c) {
      const bf16_t* p = bBase + (size_t)(c * 16) * KD + k0 + boff;
      v16bf b = load_frag(p, p + 16);
#pragma unroll
      for (int r = 0; r < 4; ++r) {
        acc[c][r] = __builtin_amdgcn_wmma_f32_16x16x32_bf16(false, a[r], false, b, (short)0,
                                                            acc[c][r], false, false);
      }
    }
  }

#pragma unroll
  for (int c = 0; c < 4; ++c) {
    float bv = bias[cbase + c * 16 + l15];
#pragma unroll
    for (int r = 0; r < 4; ++r) {
      const int rbase = r * 16 + rhalf;
#pragma unroll
      for (int i = 0; i < 8; ++i) {
        float v = silu_f(acc[c][r][i] + bv);
        outLds[(size_t)(rbase + i) * HDIM + cbase + c * 16 + l15] = (bf16_t)v;
      }
    }
  }
}

// Final layer: 64 rows x 272 (padded) cols, f32 output to LDS, bias only.
__device__ __forceinline__ void gemm_final(const bf16_t* __restrict__ inLds,
                                           const bf16_t* __restrict__ Wt,
                                           const float* __restrict__ bias,
                                           float* __restrict__ outF) {
  const int lane  = threadIdx.x & 31;
  const int wave  = threadIdx.x >> 5;
  const int l15   = lane & 15;
  const int aoff  = (lane >> 4) * 8;
  const int boff  = (lane >> 4) * 8;
  const int rhalf = (lane >> 4) * 8;

  const bf16_t* aBase = inLds + (size_t)l15 * HDIM;

  for (int c0 = wave; c0 < NPAD / 16; c0 += NTHREADS / 32) {
    const int colBase = c0 * 16;
    v8f acc[4];
#pragma unroll
    for (int r = 0; r < 4; ++r)
#pragma unroll
      for (int i = 0; i < 8; ++i) acc[r][i] = 0.0f;

    const bf16_t* bCol = Wt + (size_t)(colBase + l15) * HDIM;
    for (int k0 = 0; k0 < HDIM; k0 += 32) {
      v16bf b = load_frag(bCol + k0 + boff, bCol + k0 + boff + 16);
#pragma unroll
      for (int r = 0; r < 4; ++r) {
        const bf16_t* p = aBase + (size_t)(r * 16) * HDIM + k0 + aoff;
        v16bf a = load_frag(p, p + 16);
        acc[r] = __builtin_amdgcn_wmma_f32_16x16x32_bf16(false, a, false, b, (short)0,
                                                         acc[r], false, false);
      }
    }

    float bv = (colBase + l15 < NREAL) ? bias[colBase + l15] : 0.0f;
#pragma unroll
    for (int r = 0; r < 4; ++r) {
      const int rbase = r * 16 + rhalf;
#pragma unroll
      for (int i = 0; i < 8; ++i) {
        outF[(size_t)(rbase + i) * NPAD + colBase + l15] = acc[r][i] + bv;
      }
    }
  }
}

// LDS (196KB) already limits us to ONE 8-wave block per WGP = 2 waves/SIMD,
// so cap the occupancy target at 2 waves/EU and let the allocator use half
// the register file per wave (~200 live VGPRs needed; no spills).
__global__ void __launch_bounds__(NTHREADS, 2)
fused_mlp_kernel(const bf16_t* __restrict__ xbf,
                 const int* __restrict__ counts, const int* __restrict__ rowlist,
                 const bf16_t* __restrict__ Wt1, const float* __restrict__ b1,
                 const bf16_t* __restrict__ Wt2, const float* __restrict__ b2,
                 const bf16_t* __restrict__ Wt3, const float* __restrict__ b3,
                 const bf16_t* __restrict__ Wto, const float* __restrict__ bo,
                 const float* __restrict__ maxlv, const float* __restrict__ minlv,
                 const float* __restrict__ obs, const float* __restrict__ noise,
                 float* __restrict__ out_next, float* __restrict__ out_rew,
                 int B, int tilesPerModel) {
  extern __shared__ char smem[];
  bf16_t* bufA = (bf16_t*)smem;                                       // ROWS*HDIM bf16
  bf16_t* bufB = bufA + (size_t)ROWS * HDIM;                          // ROWS*HDIM bf16
  float*  outf = (float*)(smem + (size_t)2 * ROWS * HDIM * sizeof(bf16_t));
  int* rowIdx = (int*)(smem + (size_t)2 * ROWS * HDIM * sizeof(bf16_t)
                            + (size_t)ROWS * NPAD * sizeof(float));

  const int m    = blockIdx.x / tilesPerModel;
  const int tb   = blockIdx.x % tilesPerModel;
  const int cnt  = counts[m];
  const int base = tb * ROWS;
  if (base >= cnt) return;
  const int nActive = min(ROWS, cnt - base);
  const int tid = threadIdx.x;

  if (tid < ROWS) rowIdx[tid] = (tid < nActive) ? rowlist[(size_t)m * B + base + tid] : -1;
  __syncthreads();

  // Stage normalized-input tile into bufA, row-major ld = DIN (bf16, 16B vectors)
  for (int idx = tid; idx < ROWS * (DIN / 8); idx += NTHREADS) {
    int i = idx / (DIN / 8), seg = idx % (DIN / 8);
    int r = rowIdx[i];
    uint4 v = {0u, 0u, 0u, 0u};
    if (r >= 0) v = ((const uint4*)(xbf + (size_t)r * DIN))[seg];
    ((uint4*)bufA)[idx] = v;
  }
  __syncthreads();

  const size_t mw1 = (size_t)m * HDIM * DIN;
  const size_t mw  = (size_t)m * HDIM * HDIM;
  const size_t mwo = (size_t)m * NPAD * HDIM;

  gemm_hidden<DIN >(bufA, Wt1 + mw1, b1 + (size_t)m * HDIM, bufB);
  __syncthreads();
  gemm_hidden<HDIM>(bufB, Wt2 + mw,  b2 + (size_t)m * HDIM, bufA);
  __syncthreads();
  gemm_hidden<HDIM>(bufA, Wt3 + mw,  b3 + (size_t)m * HDIM, bufB);
  __syncthreads();
  gemm_final(bufB, Wto + mwo, bo + (size_t)m * NREAL, outf);
  __syncthreads();

  // Post-process: logvar clamp via softplus, sample, residual add, scatter out.
  for (int idx = tid; idx < ROWS * KOUT; idx += NTHREADS) {
    int i = idx / KOUT, k = idx % KOUT;
    int r = rowIdx[i];
    if (r < 0) continue;
    float mean = outf[(size_t)i * NPAD + k];
    float raw  = outf[(size_t)i * NPAD + KOUT + k];
    float mx = maxlv[k], mn = minlv[k];
    float lv = mx - softplus_f(mx - raw);
    lv = mn + softplus_f(lv - mn);
    float s = mean + noise[(size_t)r * KOUT + k] * expf(-0.5f * lv);
    if (k == 0) out_rew[r] = s;
    else out_next[(size_t)r * DIMO + (k - 1)] = s + obs[(size_t)r * DIMO + (k - 1)];
  }
}

// ---------------- launch ----------------

extern "C" void kernel_launch(void* const* d_in, const int* in_sizes, int n_in,
                              void* d_out, int out_size, void* d_ws, size_t ws_size,
                              hipStream_t stream) {
  (void)n_in; (void)out_size; (void)ws_size;
  const float* obs   = (const float*)d_in[0];
  const float* act   = (const float*)d_in[1];
  const float* mu    = (const float*)d_in[2];
  const float* sigma = (const float*)d_in[3];
  const float* W1    = (const float*)d_in[4];
  const float* b1    = (const float*)d_in[5];
  const float* W2    = (const float*)d_in[6];
  const float* b2    = (const float*)d_in[7];
  const float* W3    = (const float*)d_in[8];
  const float* b3    = (const float*)d_in[9];
  const float* Wo    = (const float*)d_in[10];
  const float* bo    = (const float*)d_in[11];
  const float* maxlv = (const float*)d_in[12];
  const float* minlv = (const float*)d_in[13];
  const int*   elite = (const int*)d_in[14];
  const int*   midx  = (const int*)d_in[15];
  const float* noise = (const float*)d_in[16];

  const int B = in_sizes[0] / DIMO;   // 32768
  const int M = in_sizes[5] / HDIM;   // 7

  char* ws = (char*)d_ws;
  size_t off = 0;
  auto take = [&](size_t bytes) {
    char* p = ws + off;
    off = (off + bytes + 255) & ~(size_t)255;
    return p;
  };
  int*    counts  = (int*)take((size_t)M * 4);
  int*    rowlist = (int*)take((size_t)M * B * 4);
  bf16_t* xbf     = (bf16_t*)take((size_t)B * DIN * 2);
  bf16_t* Wt1     = (bf16_t*)take((size_t)M * HDIM * DIN * 2);
  bf16_t* Wt2     = (bf16_t*)take((size_t)M * HDIM * HDIM * 2);
  bf16_t* Wt3     = (bf16_t*)take((size_t)M * HDIM * HDIM * 2);
  bf16_t* Wto     = (bf16_t*)take((size_t)M * NPAD * HDIM * 2);

  hipMemsetAsync(counts, 0, (size_t)M * 4, stream);

  {
    long total = (long)B * DIN;
    normalize_x_kernel<<<(int)((total + 255) / 256), 256, 0, stream>>>(obs, act, mu, sigma, xbf, B);
  }
  bin_rows_kernel<<<(B + 255) / 256, 256, 0, stream>>>(elite, midx, counts, rowlist, B);
  {
    long t1 = (long)M * HDIM * DIN;
    transpose_w_kernel<<<(int)((t1 + 255) / 256), 256, 0, stream>>>(W1, Wt1, DIN, HDIM, HDIM, t1);
    long t2 = (long)M * HDIM * HDIM;
    transpose_w_kernel<<<(int)((t2 + 255) / 256), 256, 0, stream>>>(W2, Wt2, HDIM, HDIM, HDIM, t2);
    transpose_w_kernel<<<(int)((t2 + 255) / 256), 256, 0, stream>>>(W3, Wt3, HDIM, HDIM, HDIM, t2);
    long to = (long)M * NPAD * HDIM;
    transpose_w_kernel<<<(int)((to + 255) / 256), 256, 0, stream>>>(Wo, Wto, HDIM, NREAL, NPAD, to);
  }

  const int tilesPerModel = (B + ROWS - 1) / ROWS;
  const size_t smemBytes = (size_t)2 * ROWS * HDIM * sizeof(bf16_t)   // two activation buffers
                         + (size_t)ROWS * NPAD * sizeof(float)        // f32 output tile
                         + (size_t)ROWS * sizeof(int);                // row indices
  float* out_next = (float*)d_out;
  float* out_rew  = (float*)d_out + (size_t)B * DIMO;

  fused_mlp_kernel<<<dim3(M * tilesPerModel), dim3(NTHREADS), smemBytes, stream>>>(
      xbf, counts, rowlist, Wt1, b1, Wt2, b2, Wt3, b3, Wto, bo,
      maxlv, minlv, obs, noise, out_next, out_rew, B, tilesPerModel);
}
